// SingleHeadAttention_31928786878645
// MI455X (gfx1250) — matrix operand order repaired and checked by reading, not compiled, not run on hardware
//
#include <hip/hip_runtime.h>
#include <hip/hip_bf16.h>
#include <stdint.h>
#include <math.h>

typedef __attribute__((ext_vector_type(16))) __bf16 v16bf;
typedef __attribute__((ext_vector_type(8)))  float  v8f;
typedef unsigned short u16;

__device__ __forceinline__ u16 f32_to_bf16(float f) {
    union { float f; uint32_t u; } v; v.f = f;
    uint32_t r = v.u + 0x7FFFu + ((v.u >> 16) & 1u);
    return (u16)(r >> 16);
}

// CDNA5 async copy: global -> LDS, 16 bytes per lane, tracked by ASYNCcnt.
__device__ __forceinline__ void async_b128(const u16* g, u16* l) {
    unsigned lds = (unsigned)(size_t)l;  // low 32 bits of generic LDS ptr = DS address
    asm volatile("global_load_async_to_lds_b128 %0, %1, off"
                 :: "v"(lds), "v"(g) : "memory");
}
__device__ __forceinline__ void wait_async0() {
    asm volatile("s_wait_asynccnt 0x0" ::: "memory");
}

// ---------------------------------------------------------------------------
// fp32 -> bf16 conversion, 8 elements / thread, b128 stores
// ---------------------------------------------------------------------------
__global__ void cvt_f32_to_bf16(const float* __restrict__ in,
                                u16* __restrict__ out, int n8) {
    int i = blockIdx.x * blockDim.x + threadIdx.x;
    if (i >= n8) return;
    const float4* p = (const float4*)in + (size_t)i * 2;
    float4 a = p[0], b = p[1];
    uint4 o;
    o.x = (uint32_t)f32_to_bf16(a.x) | ((uint32_t)f32_to_bf16(a.y) << 16);
    o.y = (uint32_t)f32_to_bf16(a.z) | ((uint32_t)f32_to_bf16(a.w) << 16);
    o.z = (uint32_t)f32_to_bf16(b.x) | ((uint32_t)f32_to_bf16(b.y) << 16);
    o.w = (uint32_t)f32_to_bf16(b.z) | ((uint32_t)f32_to_bf16(b.w) << 16);
    ((uint4*)out)[i] = o;
}

// ---------------------------------------------------------------------------
// Tiled WMMA GEMM: C[m][n] = alpha * sum_k A[m][k] * B[n][k]
// A: M x K bf16 row-major (lda), B: N x K bf16 row-major (ldb)
// flags bit0: C is bf16 (else f32); bit1: causal block-skip (skip n0 > m0+127)
// Block 256 threads = 8 wave32; tile 128x128x32; wave tile 32x64.
// Ping-pong LDS buffers filled by async global->LDS (ASYNCcnt pipeline),
// one barrier per K-tile, loop unrolled 2 tiles with peeled epilogue so the
// buffer select and the next-tile issue are branch-free constants.
// ---------------------------------------------------------------------------
#define BM 128
#define BN 128
#define BK 32
#define LDSS 40  // element stride (80 B, 16B-aligned, bank-conflict padding)

__launch_bounds__(256)
__global__ void gemm_bt_wmma(const u16* __restrict__ A, long long sA, int lda,
                             const u16* __restrict__ B, long long sB, int ldb,
                             void* __restrict__ Cv, long long sC, int ldc,
                             int K, float alpha, int flags) {
    const int m0 = blockIdx.y * BM;
    const int n0 = blockIdx.x * BN;
    if ((flags & 2) && (n0 > m0 + (BM - 1))) return;  // fully-masked causal block
    const int bz = blockIdx.z;
    A += (long long)bz * sA;
    B += (long long)bz * sB;

    __shared__ u16 As[2][BM * LDSS];
    __shared__ u16 Bs[2][BN * LDSS];

    const int tid   = threadIdx.x;
    const int ldRow = tid >> 1;          // 0..127
    const int ldCol = (tid & 1) * 16;    // 0 or 16 (elements)

    const int wave  = tid >> 5;
    const int lane  = tid & 31;
    const int waveM = (wave >> 1) * 32;  // 0,32,64,96
    const int waveN = (wave & 1) * 64;   // 0,64
    const int lrow  = lane & 15;
    const int half  = lane >> 4;

    const u16* gA = A + (long long)(m0 + ldRow) * lda + ldCol;
    const u16* gB = B + (long long)(n0 + ldRow) * ldb + ldCol;
    u16* lA0 = &As[0][ldRow * LDSS + ldCol];
    u16* lB0 = &Bs[0][ldRow * LDSS + ldCol];

    v8f acc[2][4] = {};
    union Fr { uint4 q[2]; v16bf v; };

    // issue one 128x32 A tile + 128x32 B tile into buffer `which`
    auto issue_tile = [&](int kk, int which) {
        const u16* nA = gA + kk;
        const u16* nB = gB + kk;
        u16* dA = lA0 + which * (BM * LDSS);
        u16* dB = lB0 + which * (BN * LDSS);
        async_b128(nA,     dA);
        async_b128(nA + 8, dA + 8);
        async_b128(nB,     dB);
        async_b128(nB + 8, dB + 8);
    };

    // consume buffer `which`: 12 ds_load_b128 fragment gathers + 8 WMMAs/wave
    auto compute_tile = [&](int which) {
        Fr af[2], bfr[4];
        // A fragment (16x32 bf16): lane half selects K groups {0..7,16..23}/{8..15,24..31}
        #pragma unroll
        for (int mt = 0; mt < 2; ++mt) {
            const u16* r = &As[which][(waveM + mt * 16 + lrow) * LDSS];
            af[mt].q[0] = *(const uint4*)(r + half * 8);
            af[mt].q[1] = *(const uint4*)(r + 16 + half * 8);
        }
        // B fragment (32x16 bf16): lane half selects K groups {0..7,16..23}/{8..15,24..31}
        #pragma unroll
        for (int nt = 0; nt < 4; ++nt) {
            const u16* r = &Bs[which][(waveN + nt * 16 + lrow) * LDSS];
            bfr[nt].q[0] = *(const uint4*)(r + half * 8);
            bfr[nt].q[1] = *(const uint4*)(r + 16 + half * 8);
        }
        #pragma unroll
        for (int mt = 0; mt < 2; ++mt)
            #pragma unroll
            for (int nt = 0; nt < 4; ++nt)
                acc[mt][nt] = __builtin_amdgcn_wmma_f32_16x16x32_bf16(
                    false, af[mt].v, false, bfr[nt].v,
                    (short)0, acc[mt][nt], false, false);
    };

    const int nt = K / BK;  // always even (K in {1024, 2048})
    issue_tile(0, 0);

    int t = 0;
    for (; t + 2 < nt; t += 2) {
        wait_async0();                    // tile t landed (this wave)
        __syncthreads();                  // all waves' fills landed; prev readers done
        issue_tile((t + 1) * BK, 1);
        compute_tile(0);
        wait_async0();                    // tile t+1 landed
        __syncthreads();
        issue_tile((t + 2) * BK, 0);
        compute_tile(1);
    }
    // epilogue: tiles nt-2 (buf0) and nt-1 (buf1)
    wait_async0();
    __syncthreads();
    issue_tile((nt - 1) * BK, 1);
    compute_tile(0);
    wait_async0();
    __syncthreads();
    compute_tile(1);

    // ---- store: lane = N, VGPR r = M (+8 for upper lane half) ----
    const bool outbf = flags & 1;
    #pragma unroll
    for (int mt = 0; mt < 2; ++mt) {
        #pragma unroll
        for (int nt2 = 0; nt2 < 4; ++nt2) {
            const int n     = n0 + waveN + nt2 * 16 + lrow;
            const int mbase = m0 + waveM + mt * 16 + half * 8;
            if (outbf) {
                u16* C = (u16*)Cv + (long long)bz * sC;
                #pragma unroll
                for (int r = 0; r < 8; ++r)
                    C[(long long)(mbase + r) * ldc + n] = f32_to_bf16(acc[mt][nt2][r] * alpha);
            } else {
                float* C = (float*)Cv + (long long)bz * sC;
                #pragma unroll
                for (int r = 0; r < 8; ++r)
                    C[(long long)(mbase + r) * ldc + n] = acc[mt][nt2][r] * alpha;
            }
        }
    }
}

// ---------------------------------------------------------------------------
// bf16 transpose (B,T,D) -> (B,D,T) via padded LDS tile
// ---------------------------------------------------------------------------
__global__ void transpose_bf16(const u16* __restrict__ in, u16* __restrict__ out,
                               int T, int D) {
    __shared__ u16 tile[32][33];
    const int b  = blockIdx.z;
    const int d0 = blockIdx.x * 32;
    const int t0 = blockIdx.y * 32;
    const u16* src = in  + (long long)b * T * D;
    u16*       dst = out + (long long)b * D * T;
    const int x = threadIdx.x;   // 0..31
    const int y = threadIdx.y;   // 0..7
    #pragma unroll
    for (int k = 0; k < 4; ++k)
        tile[y + 8 * k][x] = src[(long long)(t0 + y + 8 * k) * D + d0 + x];
    __syncthreads();
    #pragma unroll
    for (int k = 0; k < 4; ++k)
        dst[(long long)(d0 + y + 8 * k) * T + t0 + x] = tile[x][y + 8 * k];
}

// ---------------------------------------------------------------------------
// Causal row softmax over S (f32, one block per row), written back IN PLACE
// as packed bf16 (masked cols = 0, so the following dense PV GEMM is exact).
// ---------------------------------------------------------------------------
__launch_bounds__(256)
__global__ void softmax_causal_inplace(float* __restrict__ S, int T) {
    const int t = blockIdx.x;
    const int b = blockIdx.y;
    float* row = S + ((long long)b * T + t) * (long long)T;

    const int tid  = threadIdx.x;
    const int wave = tid >> 5;
    const int lane = tid & 31;

    float v[8];
    float mx = -INFINITY;
    #pragma unroll
    for (int k = 0; k < 8; ++k) {
        const int j = tid + k * 256;
        v[k] = (j <= t) ? row[j] : -INFINITY;
        mx = fmaxf(mx, v[k]);
    }
    #pragma unroll
    for (int o = 16; o > 0; o >>= 1) mx = fmaxf(mx, __shfl_xor(mx, o, 32));

    __shared__ float red[8];
    if (lane == 0) red[wave] = mx;
    __syncthreads();
    if (wave == 0) {
        float m2 = (lane < 8) ? red[lane] : -INFINITY;
        #pragma unroll
        for (int o = 4; o > 0; o >>= 1) m2 = fmaxf(m2, __shfl_xor(m2, o, 32));
        if (lane == 0) red[0] = m2;
    }
    __syncthreads();
    mx = red[0];

    float e[8], s = 0.f;
    #pragma unroll
    for (int k = 0; k < 8; ++k) { e[k] = __expf(v[k] - mx); s += e[k]; }
    #pragma unroll
    for (int o = 16; o > 0; o >>= 1) s += __shfl_xor(s, o, 32);
    __syncthreads();
    if (lane == 0) red[wave] = s;
    __syncthreads();
    if (wave == 0) {
        float s2 = (lane < 8) ? red[lane] : 0.f;
        #pragma unroll
        for (int o = 4; o > 0; o >>= 1) s2 += __shfl_xor(s2, o, 32);
        if (lane == 0) red[0] = s2;
    }
    __syncthreads();
    const float inv = 1.0f / red[0];

    u16* prow = (u16*)row;  // pack bf16 P into the front of the f32 row
    #pragma unroll
    for (int k = 0; k < 8; ++k) {
        const int j = tid + k * 256;
        prow[j] = f32_to_bf16(e[k] * inv);
    }
}

// ---------------------------------------------------------------------------
extern "C" void kernel_launch(void* const* d_in, const int* in_sizes, int n_in,
                              void* d_out, int out_size, void* d_ws, size_t ws_size,
                              hipStream_t stream) {
    const int B = 4, T = 2048, D = 1024;
    const long long MT = (long long)B * T;  // 8192 total rows

    const float* x  = (const float*)d_in[0];
    const float* Wq = (const float*)d_in[1];
    const float* Wk = (const float*)d_in[2];
    const float* Wv = (const float*)d_in[3];
    const float* Wo = (const float*)d_in[4];

    char* ws = (char*)d_ws;
    size_t off = 0;
    auto alloc = [&](size_t bytes) -> void* {
        void* p = ws + off;
        off += (bytes + 255) & ~(size_t)255;
        return p;
    };
    u16*   Xb  = (u16*)  alloc((size_t)MT * D * 2);
    u16*   Wqb = (u16*)  alloc((size_t)D * D * 2);
    u16*   Wkb = (u16*)  alloc((size_t)D * D * 2);
    u16*   Wvb = (u16*)  alloc((size_t)D * D * 2);
    u16*   Wob = (u16*)  alloc((size_t)D * D * 2);
    u16*   Qb  = (u16*)  alloc((size_t)MT * D * 2);
    u16*   Kb  = (u16*)  alloc((size_t)MT * D * 2);
    u16*   Vb  = (u16*)  alloc((size_t)MT * D * 2);
    u16*   Vt  = (u16*)  alloc((size_t)MT * D * 2);          // (B, D, T)
    float* S   = (float*)alloc((size_t)B * T * T * 4);       // scores / P (bf16 in place)
    u16*   Ob  = (u16*)  alloc((size_t)MT * D * 2);

    // 1) fp32 -> bf16 conversions
    cvt_f32_to_bf16<<<(int)(MT * D / 8 / 256), 256, 0, stream>>>(x, Xb, (int)(MT * D / 8));
    cvt_f32_to_bf16<<<D * D / 8 / 256, 256, 0, stream>>>(Wq, Wqb, D * D / 8);
    cvt_f32_to_bf16<<<D * D / 8 / 256, 256, 0, stream>>>(Wk, Wkb, D * D / 8);
    cvt_f32_to_bf16<<<D * D / 8 / 256, 256, 0, stream>>>(Wv, Wvb, D * D / 8);
    cvt_f32_to_bf16<<<D * D / 8 / 256, 256, 0, stream>>>(Wo, Wob, D * D / 8);

    // 2) Q/K/V projections: (8192x1024) @ (1024x1024)^T -> bf16
    dim3 gProj(D / BN, (unsigned)(MT / BM), 1);
    gemm_bt_wmma<<<gProj, 256, 0, stream>>>(Xb, 0, D, Wqb, 0, D, Qb, 0, D, D, 1.0f, 1);
    gemm_bt_wmma<<<gProj, 256, 0, stream>>>(Xb, 0, D, Wkb, 0, D, Kb, 0, D, D, 1.0f, 1);
    gemm_bt_wmma<<<gProj, 256, 0, stream>>>(Xb, 0, D, Wvb, 0, D, Vb, 0, D, D, 1.0f, 1);

    // 3) V -> V^T per batch
    transpose_bf16<<<dim3(D / 32, T / 32, B), dim3(32, 8), 0, stream>>>(Vb, Vt, T, D);

    // 4) S = (Q K^T) / sqrt(D), f32, causal block-skip
    gemm_bt_wmma<<<dim3(T / BN, T / BM, B), 256, 0, stream>>>(
        Qb, (long long)T * D, D, Kb, (long long)T * D, D,
        S, (long long)T * T, T, D, 1.0f / 32.0f, 2);

    // 5) causal softmax -> P (bf16, in place, lda = 2T as bf16 elements)
    softmax_causal_inplace<<<dim3(T, B), 256, 0, stream>>>(S, T);

    // 6) O = P @ V  (A = packed bf16 P, lda = 2T; B = V^T rows = D x T)
    gemm_bt_wmma<<<dim3(D / BN, T / BM, B), 256, 0, stream>>>(
        (const u16*)S, (long long)T * T * 2, 2 * T,
        Vt, (long long)D * T, T,
        Ob, (long long)T * D, D, T, 1.0f, 1);

    // 7) Out = O @ W_o^T -> fp32 d_out
    gemm_bt_wmma<<<dim3(D / BN, (unsigned)(MT / BM), 1), 256, 0, stream>>>(
        Ob, 0, D, Wob, 0, D, d_out, 0, D, D, 1.0f, 0);
}
